// SchNet_7344394076177
// MI455X (gfx1250) — matrix operand order repaired and checked by reading, not compiled, not run on hardware
//
#include <hip/hip_runtime.h>
#include <hip/hip_bf16.h>
#include <cstddef>

#define NATOM 20000
#define NEDGE 320000
#define HID   128
#define NG    128
#define NLAY  4
#define NBAT  200
#define PI_F  3.14159265358979f
#define CUTF  5.0f
#define GSTEP (CUTF / 127.0f)
#define GCOEF (-0.5f / (GSTEP * GSTEP))
#define LOG2F_ 0.6931471805599453f

typedef __attribute__((ext_vector_type(16))) __bf16 v16bf;
typedef __attribute__((ext_vector_type(8)))  float  v8f;

__device__ __forceinline__ float sspf(float x) {
  return fmaxf(x, 0.f) + log1pf(expf(-fabsf(x))) - LOG2F_;
}
__device__ __forceinline__ float sigf(float x) { return 1.f / (1.f + expf(-x)); }
__device__ __forceinline__ void atomAddF(float* p, float v) {
  __hip_atomic_fetch_add(p, v, __ATOMIC_RELAXED, __HIP_MEMORY_SCOPE_AGENT);
}
__device__ __forceinline__ float halfReduce16(float v) {
  v += __shfl_xor(v, 1);
  v += __shfl_xor(v, 2);
  v += __shfl_xor(v, 4);
  v += __shfl_xor(v, 8);
  return v;
}

#define EPI_PLAIN     0
#define EPI_SSP       1
#define EPI_STORE2    2
#define EPI_SSPSIG    3
#define EPI_SCATTER   4
#define EPI_ADDRES    5
#define EPI_MULSIG    6
#define EPI_MULAUX    7
#define EPI_BWDEDGE   8
#define EPI_GDRED     9

struct EpiArgs {
  const int*   row;
  const int*   col;
  const float* Ccut;
  const float* hx;
  float*       agg;
  const float* gagg;
  float*       gF;
  float*       ghx;
  float*       gCcut;
  const float* ea;
  const float* dist;
  float*       gd;
};

// Pack f32 weight [Ke x Ne] (optionally transposed) into bf16 B-fragment layout:
// tile(nt,kc), lane l, slot j -> B_eff[kc*32 + (l>>4)*8 + (j&7) + (j>>3)*16][nt*16 + (l&15)]
__global__ void pack_b_k(const float* __restrict__ src, __bf16* __restrict__ dst,
                         int KC, int NT, int ld, int trans) {
  int idx = blockIdx.x * blockDim.x + threadIdx.x;
  int total = NT * KC * 512;
  if (idx >= total) return;
  int j  = idx & 15;
  int l  = (idx >> 4) & 31;
  int rest = idx >> 9;
  int kc = rest % KC;
  int nt = rest / KC;
  int n = nt * 16 + (l & 15);
  int k = kc * 32 + (l >> 4) * 8 + (j & 7) + (j >> 3) * 16;
  float v = trans ? src[(size_t)n * ld + k] : src[(size_t)k * ld + n];
  dst[idx] = (__bf16)v;
}

// WMMA GEMM with fused epilogues. 4 waves/block, one 16-row M tile per wave.
template <int NT, int KC, int EPI>
__global__ __launch_bounds__(128) void gemm_wmma(
    const float* __restrict__ A, const __bf16* __restrict__ Bp,
    const float* __restrict__ bias, float* __restrict__ out1,
    float* __restrict__ out2, const float* __restrict__ aux,
    const float* __restrict__ res, int M, EpiArgs ep) {
  const int lane  = threadIdx.x & 31;
  const int wave  = threadIdx.x >> 5;
  const int mtile = blockIdx.x * 4 + wave;
  if (mtile * 16 >= M) return;  // wave-uniform: EXEC all-1 for the WMMAs below
  const int hv  = lane >> 4;
  const int l15 = lane & 15;
  const int K   = KC * 32;

  const float* arow = A + (size_t)(mtile * 16 + l15) * K;
  const v16bf* bp   = (const v16bf*)Bp;

  v8f acc[NT];
#pragma unroll
  for (int nt = 0; nt < NT; ++nt)
    acc[nt] = (v8f){0.f, 0.f, 0.f, 0.f, 0.f, 0.f, 0.f, 0.f};

#pragma unroll
  for (int kc = 0; kc < KC; ++kc) {
    v16bf af;  // ISA 16-bit A 16x32 layout: two contiguous 8-element runs
    const float* ap = arow + kc * 32 + hv * 8;
#pragma unroll
    for (int j = 0; j < 8; ++j) af[j] = (__bf16)ap[j];
#pragma unroll
    for (int j = 0; j < 8; ++j) af[8 + j] = (__bf16)ap[16 + j];
#pragma unroll
    for (int nt = 0; nt < NT; ++nt) {
      v16bf bf = bp[(nt * KC + kc) * 32 + lane];
      acc[nt] = __builtin_amdgcn_wmma_f32_16x16x32_bf16(
          false, af, false, bf, (short)0, acc[nt], false, false);
    }
  }

  const int ldc = NT * 16;
  float bb[NT];
#pragma unroll
  for (int nt = 0; nt < NT; ++nt) bb[nt] = bias ? bias[nt * 16 + l15] : 0.f;

  const int rbase = mtile * 16 + hv * 8;
#pragma unroll
  for (int r = 0; r < 8; ++r) {
    const int m = rbase + r;

    if (EPI == EPI_PLAIN) {
#pragma unroll
      for (int nt = 0; nt < NT; ++nt)
        out1[(size_t)m * ldc + nt * 16 + l15] = acc[nt][r] + bb[nt];
    } else if (EPI == EPI_SSP) {
#pragma unroll
      for (int nt = 0; nt < NT; ++nt)
        out1[(size_t)m * ldc + nt * 16 + l15] = sspf(acc[nt][r] + bb[nt]);
    } else if (EPI == EPI_STORE2) {
#pragma unroll
      for (int nt = 0; nt < NT; ++nt) {
        size_t o = (size_t)m * ldc + nt * 16 + l15;
        float pre = acc[nt][r] + bb[nt];
        out1[o] = pre;
        out2[o] = sspf(pre);
      }
    } else if (EPI == EPI_SSPSIG) {
#pragma unroll
      for (int nt = 0; nt < NT; ++nt) {
        size_t o = (size_t)m * ldc + nt * 16 + l15;
        float pre = acc[nt][r] + bb[nt];
        out1[o] = sspf(pre);
        out2[o] = sigf(pre);
      }
    } else if (EPI == EPI_SCATTER) {
      const int e = m;
      const float cc = ep.Ccut[e];
      const int re = ep.row[e], ce = ep.col[e];
#pragma unroll
      for (int nt = 0; nt < NT; ++nt) {
        int n = nt * 16 + l15;
        float W = (acc[nt][r] + bb[nt]) * cc;
        atomAddF(&ep.agg[(size_t)ce * HID + n], ep.hx[(size_t)re * HID + n] * W);
      }
    } else if (EPI == EPI_ADDRES) {
#pragma unroll
      for (int nt = 0; nt < NT; ++nt) {
        size_t o = (size_t)m * ldc + nt * 16 + l15;
        out1[o] = acc[nt][r] + bb[nt] + res[o];
      }
    } else if (EPI == EPI_MULSIG) {
#pragma unroll
      for (int nt = 0; nt < NT; ++nt) {
        size_t o = (size_t)m * ldc + nt * 16 + l15;
        out1[o] = (acc[nt][r] + bb[nt]) * sigf(aux[o]);
      }
    } else if (EPI == EPI_MULAUX) {
#pragma unroll
      for (int nt = 0; nt < NT; ++nt) {
        size_t o = (size_t)m * ldc + nt * 16 + l15;
        out1[o] = (acc[nt][r] + bb[nt]) * aux[o];
      }
    } else if (EPI == EPI_BWDEDGE) {
      const int e = m;
      const float cc = ep.Ccut[e];
      const int re = ep.row[e], ce = ep.col[e];
      float partial = 0.f;
#pragma unroll
      for (int nt = 0; nt < NT; ++nt) {
        int n = nt * 16 + l15;
        float F   = acc[nt][r] + bb[nt];
        float W   = F * cc;
        float gm  = ep.gagg[(size_t)ce * HID + n];
        float hxv = ep.hx[(size_t)re * HID + n];
        float gW  = gm * hxv;
        ep.gF[(size_t)e * HID + n] = gW * cc;
        atomAddF(&ep.ghx[(size_t)re * HID + n], gm * W);
        partial += gW * F;
      }
      partial = halfReduce16(partial);
      if (l15 == 0) atomAddF(&ep.gCcut[e], partial);
    } else if (EPI == EPI_GDRED) {
      const int e = m;
      const float dd = ep.dist[e];
      float partial = 0.f;
#pragma unroll
      for (int nt = 0; nt < NT; ++nt) {
        int n = nt * 16 + l15;
        float gea = acc[nt][r] + bb[nt];
        float eav = ep.ea[(size_t)e * NG + n];
        float off = n * GSTEP;
        partial += gea * eav * (2.f * GCOEF) * (dd - off);
      }
      partial = halfReduce16(partial);
      if (l15 == 0) atomAddF(&ep.gd[e], partial);
    }
  }
}

__global__ void edge_geom_k(const float* __restrict__ pos, const int* __restrict__ row,
                            const int* __restrict__ col, float* __restrict__ dist,
                            float* __restrict__ ccut, float* __restrict__ ea) {
  int idx = blockIdx.x * blockDim.x + threadIdx.x;
  if (idx >= NEDGE * NG) return;
  int e = idx >> 7, g = idx & 127;
  int r = row[e], c = col[e];
  float dx = pos[r * 3 + 0] - pos[c * 3 + 0];
  float dy = pos[r * 3 + 1] - pos[c * 3 + 1];
  float dz = pos[r * 3 + 2] - pos[c * 3 + 2];
  float d = sqrtf(dx * dx + dy * dy + dz * dz);
  float off = g * GSTEP;
  float t = d - off;
  ea[idx] = expf(GCOEF * t * t);
  if (g == 0) {
    dist[e] = d;
    ccut[e] = 0.5f * (cosf(d * (PI_F / CUTF)) + 1.f);
  }
}

__global__ void embed_k(const float* __restrict__ emb, const int* __restrict__ z,
                        float* __restrict__ h0) {
  int idx = blockIdx.x * blockDim.x + threadIdx.x;
  if (idx >= NATOM * HID) return;
  int i = idx >> 7, n = idx & 127;
  h0[idx] = emb[(size_t)z[i] * HID + n];
}

__global__ void head_final_k(const float* __restrict__ u2, const float* __restrict__ w2,
                             const float* __restrict__ b2, const int* __restrict__ batch,
                             float* __restrict__ out) {
  int i = blockIdx.x * blockDim.x + threadIdx.x;
  if (i >= NATOM) return;
  const float* u = u2 + (size_t)i * 64;
  float s = b2[0];
  for (int j = 0; j < 64; ++j) s += u[j] * w2[j];
  atomAddF(&out[batch[i]], s);
}

__global__ void head_grad_k(const float* __restrict__ aH, const float* __restrict__ w2,
                            float* __restrict__ gaH) {
  int idx = blockIdx.x * blockDim.x + threadIdx.x;
  if (idx >= NATOM * 64) return;
  int j = idx & 63;
  gaH[idx] = sigf(aH[idx]) * w2[j];
}

__global__ void force_k(const float* __restrict__ pos, const int* __restrict__ row,
                        const int* __restrict__ col, const float* __restrict__ dist,
                        const float* __restrict__ gd, const float* __restrict__ gcc,
                        float* __restrict__ force) {
  int e = blockIdx.x * blockDim.x + threadIdx.x;
  if (e >= NEDGE) return;
  int r = row[e], c = col[e];
  float dx = pos[r * 3 + 0] - pos[c * 3 + 0];
  float dy = pos[r * 3 + 1] - pos[c * 3 + 1];
  float dz = pos[r * 3 + 2] - pos[c * 3 + 2];
  float d = dist[e];
  float g = gd[e] + gcc[e] * (-0.5f * (PI_F / CUTF)) * sinf(d * (PI_F / CUTF));
  float s = g / d;
  float fx = s * dx, fy = s * dy, fz = s * dz;
  atomAddF(&force[r * 3 + 0], -fx);
  atomAddF(&force[r * 3 + 1], -fy);
  atomAddF(&force[r * 3 + 2], -fz);
  atomAddF(&force[c * 3 + 0], fx);
  atomAddF(&force[c * 3 + 1], fy);
  atomAddF(&force[c * 3 + 2], fz);
}

template <int NT, int KC, int EPI>
static inline void G(const float* A, const __bf16* Bp, const float* bias, float* o1,
                     float* o2, const float* aux, const float* res, int M,
                     const EpiArgs& ep, hipStream_t s) {
  int mt = (M + 15) / 16;
  gemm_wmma<NT, KC, EPI><<<dim3((mt + 3) / 4), dim3(128), 0, s>>>(
      A, Bp, bias, o1, o2, aux, res, M, ep);
}

extern "C" void kernel_launch(void* const* d_in, const int* in_sizes, int n_in,
                              void* d_out, int out_size, void* d_ws, size_t ws_size,
                              hipStream_t stream) {
  const float* pos    = (const float*)d_in[0];
  const int*   z      = (const int*)d_in[1];
  const int*   batch  = (const int*)d_in[2];
  const int*   eidx   = (const int*)d_in[3];
  const float* emb    = (const float*)d_in[4];
  const float* mlp_w1 = (const float*)d_in[5];
  const float* mlp_b1 = (const float*)d_in[6];
  const float* mlp_w2 = (const float*)d_in[7];
  const float* mlp_b2 = (const float*)d_in[8];
  const float* lin1w  = (const float*)d_in[9];
  const float* lin2w  = (const float*)d_in[10];
  const float* lin2b  = (const float*)d_in[11];
  const float* blkw   = (const float*)d_in[12];
  const float* blkb   = (const float*)d_in[13];
  const float* hw1    = (const float*)d_in[14];
  const float* hb1    = (const float*)d_in[15];
  const float* hw2    = (const float*)d_in[16];
  const float* hb2    = (const float*)d_in[17];
  (void)n_in; (void)in_sizes; (void)out_size; (void)ws_size;

  const int* row = eidx;
  const int* col = eidx + NEDGE;
  float* out = (float*)d_out;    // [NBAT energies | NATOM*3 forces]
  float* force = out + NBAT;

  size_t off = 0;
  auto alloc = [&](size_t bytes) -> void* {
    off = (off + 255) & ~(size_t)255;
    void* p = (char*)d_ws + off;
    off += bytes;
    return p;
  };
  const size_t NH = (size_t)NATOM * HID;
  const size_t EH = (size_t)NEDGE * HID;

  __bf16 *w1p[NLAY], *w2p[NLAY], *l1p[NLAY], *l2p[NLAY], *bkp[NLAY];
  __bf16 *w1tp[NLAY], *w2tp[NLAY], *l1tp[NLAY], *l2tp[NLAY], *bktp[NLAY];
  for (int i = 0; i < NLAY; ++i) {
    w1p[i]  = (__bf16*)alloc(128 * 128 * 2);
    w2p[i]  = (__bf16*)alloc(128 * 128 * 2);
    l1p[i]  = (__bf16*)alloc(128 * 128 * 2);
    l2p[i]  = (__bf16*)alloc(128 * 128 * 2);
    bkp[i]  = (__bf16*)alloc(128 * 128 * 2);
    w1tp[i] = (__bf16*)alloc(128 * 128 * 2);
    w2tp[i] = (__bf16*)alloc(128 * 128 * 2);
    l1tp[i] = (__bf16*)alloc(128 * 128 * 2);
    l2tp[i] = (__bf16*)alloc(128 * 128 * 2);
    bktp[i] = (__bf16*)alloc(128 * 128 * 2);
  }
  __bf16* hw1p  = (__bf16*)alloc(128 * 64 * 2);
  __bf16* hw1tp = (__bf16*)alloc(64 * 128 * 2);

  float* dist = (float*)alloc(NEDGE * 4);
  float* ccut = (float*)alloc(NEDGE * 4);
  float* gd   = (float*)alloc(NEDGE * 4);
  float* gcc  = (float*)alloc(NEDGE * 4);
  float* ea   = (float*)alloc((size_t)NEDGE * NG * 4);
  float* hst  = (float*)alloc(5 * NH * 4);
  float* hxst = (float*)alloc(4 * NH * 4);
  float* vst  = (float*)alloc(4 * NH * 4);
  float* aH   = (float*)alloc((size_t)NATOM * 64 * 4);
  float* u2   = (float*)alloc((size_t)NATOM * 64 * 4);
  float* gaH  = (float*)alloc((size_t)NATOM * 64 * 4);
  float* nb0  = (float*)alloc(NH * 4);
  float* nb1  = (float*)alloc(NH * 4);
  float* nb2  = (float*)alloc(NH * 4);
  float* nb3  = (float*)alloc(NH * 4);
  float* S    = (float*)alloc(EH * 4);
  float* T1   = (float*)alloc(EH * 4);
  float* C1   = (float*)alloc(EH * 4);

  EpiArgs ez{};

  auto packL = [&](const float* src, __bf16* dst, int KC, int NT, int ld, int tr) {
    int total = NT * KC * 512;
    pack_b_k<<<dim3((total + 255) / 256), dim3(256), 0, stream>>>(src, dst, KC, NT, ld, tr);
  };
  for (int i = 0; i < NLAY; ++i) {
    const size_t HH = (size_t)HID * HID;
    packL(mlp_w1 + i * (size_t)NG * HID, w1p[i], 4, 8, 128, 0);
    packL(mlp_w2 + i * HH, w2p[i], 4, 8, 128, 0);
    packL(lin1w + i * HH, l1p[i], 4, 8, 128, 0);
    packL(lin2w + i * HH, l2p[i], 4, 8, 128, 0);
    packL(blkw + i * HH, bkp[i], 4, 8, 128, 0);
    packL(mlp_w1 + i * (size_t)NG * HID, w1tp[i], 4, 8, 128, 1);
    packL(mlp_w2 + i * HH, w2tp[i], 4, 8, 128, 1);
    packL(lin1w + i * HH, l1tp[i], 4, 8, 128, 1);
    packL(lin2w + i * HH, l2tp[i], 4, 8, 128, 1);
    packL(blkw + i * HH, bktp[i], 4, 8, 128, 1);
  }
  packL(hw1, hw1p, 4, 4, 64, 0);
  packL(hw1, hw1tp, 2, 8, 64, 1);

  hipMemsetAsync(d_out, 0, (NBAT + (size_t)NATOM * 3) * 4, stream);
  hipMemsetAsync(gd, 0, NEDGE * 4, stream);
  hipMemsetAsync(gcc, 0, NEDGE * 4, stream);

  {
    int tot = NEDGE * NG;
    edge_geom_k<<<dim3((tot + 255) / 256), dim3(256), 0, stream>>>(pos, row, col, dist, ccut, ea);
    tot = NATOM * HID;
    embed_k<<<dim3((tot + 255) / 256), dim3(256), 0, stream>>>(emb, z, hst);
  }

  // ---------------- forward ----------------
  for (int i = 0; i < NLAY; ++i) {
    float* hin  = hst + (size_t)i * NH;
    float* hout = hst + (size_t)(i + 1) * NH;
    float* hx   = hxst + (size_t)i * NH;
    float* v    = vst + (size_t)i * NH;
    G<8, 4, EPI_PLAIN>(hin, l1p[i], nullptr, hx, nullptr, nullptr, nullptr, NATOM, ez, stream);
    G<8, 4, EPI_SSP>(ea, w1p[i], mlp_b1 + i * HID, S, nullptr, nullptr, nullptr, NEDGE, ez, stream);
    hipMemsetAsync(nb0, 0, NH * 4, stream);
    {
      EpiArgs ep = ez;
      ep.row = row; ep.col = col; ep.Ccut = ccut; ep.hx = hx; ep.agg = nb0;
      G<8, 4, EPI_SCATTER>(S, w2p[i], mlp_b2 + i * HID, nullptr, nullptr, nullptr, nullptr,
                           NEDGE, ep, stream);
    }
    G<8, 4, EPI_STORE2>(nb0, l2p[i], lin2b + i * HID, v, nb1, nullptr, nullptr, NATOM, ez, stream);
    G<8, 4, EPI_ADDRES>(nb1, bkp[i], blkb + i * HID, hout, nullptr, nullptr, hin, NATOM, ez, stream);
  }

  float* h4 = hst + (size_t)NLAY * NH;
  G<4, 4, EPI_STORE2>(h4, hw1p, hb1, aH, u2, nullptr, nullptr, NATOM, ez, stream);
  head_final_k<<<dim3((NATOM + 255) / 256), dim3(256), 0, stream>>>(u2, hw2, hb2, batch, out);

  // ---------------- backward ----------------
  {
    int tot = NATOM * 64;
    head_grad_k<<<dim3((tot + 255) / 256), dim3(256), 0, stream>>>(aH, hw2, gaH);
  }
  G<8, 2, EPI_PLAIN>(gaH, hw1tp, nullptr, nb3, nullptr, nullptr, nullptr, NATOM, ez, stream);

  for (int i = NLAY - 1; i >= 0; --i) {
    float* hx = hxst + (size_t)i * NH;
    float* v  = vst + (size_t)i * NH;
    G<8, 4, EPI_MULSIG>(nb3, bktp[i], nullptr, nb1, nullptr, v, nullptr, NATOM, ez, stream);
    G<8, 4, EPI_PLAIN>(nb1, l2tp[i], nullptr, nb0, nullptr, nullptr, nullptr, NATOM, ez, stream);
    G<8, 4, EPI_SSPSIG>(ea, w1p[i], mlp_b1 + i * HID, S, T1, nullptr, nullptr, NEDGE, ez, stream);
    hipMemsetAsync(nb2, 0, NH * 4, stream);
    {
      EpiArgs ep = ez;
      ep.row = row; ep.col = col; ep.Ccut = ccut; ep.hx = hx;
      ep.gagg = nb0; ep.gF = C1; ep.ghx = nb2; ep.gCcut = gcc;
      G<8, 4, EPI_BWDEDGE>(S, w2p[i], mlp_b2 + i * HID, nullptr, nullptr, nullptr, nullptr,
                           NEDGE, ep, stream);
    }
    G<8, 4, EPI_MULAUX>(C1, w2tp[i], nullptr, S, nullptr, T1, nullptr, NEDGE, ez, stream);
    {
      EpiArgs ep = ez;
      ep.ea = ea; ep.dist = dist; ep.gd = gd;
      G<8, 4, EPI_GDRED>(S, w1tp[i], nullptr, nullptr, nullptr, nullptr, nullptr, NEDGE, ep, stream);
    }
    G<8, 4, EPI_ADDRES>(nb2, l1tp[i], nullptr, nb3, nullptr, nullptr, nb3, NATOM, ez, stream);
  }

  force_k<<<dim3((NEDGE + 255) / 256), dim3(256), 0, stream>>>(pos, row, col, dist, gd, gcc,
                                                               force);
}
